// HybridKernelMethod_65481071399132
// MI455X (gfx1250) — matrix operand order repaired and checked by reading, not compiled, not run on hardware
//
#include <hip/hip_runtime.h>
#include <hip/hip_bf16.h>

typedef __attribute__((ext_vector_type(16))) _Float16 v16h;
typedef __attribute__((ext_vector_type(8)))  _Float16 v8h;
typedef __attribute__((ext_vector_type(8)))  float    v8f;

#define EMBED 256
#define BM 128
#define BN 128
#define KC 128          // K-chunk staged in LDS (2 chunks of 128)
#define PITCH 136       // 128 + 8 halfs pad: 272B row stride, 16B aligned, conflict-free

// ---------------------------------------------------------------------------
// Row squared-norm kernel: one wave (32 lanes) per row of x [rows x 256] f32.
// ---------------------------------------------------------------------------
__global__ __launch_bounds__(256) void rownorm_kernel(const float* __restrict__ x,
                                                      float* __restrict__ out,
                                                      int rows) {
    const int lane = threadIdx.x & 31;
    const int wv   = threadIdx.x >> 5;
    const int row  = blockIdx.x * 8 + wv;
    if (row >= rows) return;
    const float* p = x + (size_t)row * EMBED + lane * 8;
    float4 u = *(const float4*)(p);
    float4 v = *(const float4*)(p + 4);
    float s = u.x*u.x + u.y*u.y + u.z*u.z + u.w*u.w
            + v.x*v.x + v.y*v.y + v.z*v.z + v.w*v.w;
#pragma unroll
    for (int off = 16; off > 0; off >>= 1) s += __shfl_xor(s, off);
    if (lane == 0) out[row] = s;
}

// ---------------------------------------------------------------------------
// Helpers
// ---------------------------------------------------------------------------
static __device__ __forceinline__ v16h ld_frag(const _Float16* p0, const _Float16* p1) {
    v8h a = *(const v8h*)p0;
    v8h b = *(const v8h*)p1;
    v16h r;
#pragma unroll
    for (int i = 0; i < 8; ++i) { r[i] = a[i]; r[8 + i] = b[i]; }
    return r;
}

static __device__ __forceinline__ void split_store(_Float16* hi, _Float16* lo, float x) {
    _Float16 h = (_Float16)x;
    *hi = h;
    *lo = (_Float16)(x - (float)h);
}

// ---------------------------------------------------------------------------
// Main fused kernel: out[i,j] = 0.5*exp(-(na+nb-2ab)) + 0.5*ab
// 128x128 tile per workgroup (8 waves, wave tile 32x64). K staged in two
// 128-deep LDS chunks (139KB -> 2 workgroups per 320KB WGP).
// Split-f16 GEMM: ab = ah*bh + ah*bl + al*bh (f32 accumulate, ~fp32 accuracy)
// ---------------------------------------------------------------------------
__global__ __launch_bounds__(256) void hybrid_gram_kernel(const float* __restrict__ A,
                                                          const float* __restrict__ B,
                                                          const float* __restrict__ sqa,
                                                          const float* __restrict__ sqb,
                                                          float* __restrict__ out,
                                                          int nrows, int mcols) {
    __shared__ __align__(16) _Float16 sAhi[BM * PITCH];
    __shared__ __align__(16) _Float16 sAlo[BM * PITCH];
    __shared__ __align__(16) _Float16 sBhi[BN * PITCH];
    __shared__ __align__(16) _Float16 sBlo[BN * PITCH];

    const int tid   = threadIdx.x;
    const int rowA0 = blockIdx.y * BM;
    const int rowB0 = blockIdx.x * BN;

    const int lane = tid & 31;
    const int wv   = tid >> 5;
    const int wm   = (wv & 3) * 32;   // wave M offset (4 waves along M)
    const int wn   = (wv >> 2) * 64;  // wave N offset (2 waves along N)
    const int l16  = lane & 15;
    const int lhi  = lane >> 4;

    v8f acc[2][4];
#pragma unroll
    for (int tm = 0; tm < 2; ++tm)
#pragma unroll
        for (int tn = 0; tn < 4; ++tn)
            acc[tm][tn] = (v8f){0.f, 0.f, 0.f, 0.f, 0.f, 0.f, 0.f, 0.f};

    for (int ks = 0; ks < EMBED; ks += KC) {
        if (ks) __syncthreads();   // protect LDS reuse across chunks

        // ---- stage chunk: f32 global -> hi/lo f16 in LDS ----
#pragma unroll 4
        for (int idx = tid; idx < BM * (KC / 4); idx += 256) {
            const int r  = idx >> 5;          // row within tile (32 float4 per row)
            const int c  = (idx & 31) * 4;    // column within chunk
            float4 va = *(const float4*)(A + (size_t)(rowA0 + r) * EMBED + ks + c);
            float4 vb = *(const float4*)(B + (size_t)(rowB0 + r) * EMBED + ks + c);
            const int base = r * PITCH + c;
            split_store(&sAhi[base + 0], &sAlo[base + 0], va.x);
            split_store(&sAhi[base + 1], &sAlo[base + 1], va.y);
            split_store(&sAhi[base + 2], &sAlo[base + 2], va.z);
            split_store(&sAhi[base + 3], &sAlo[base + 3], va.w);
            split_store(&sBhi[base + 0], &sBlo[base + 0], vb.x);
            split_store(&sBhi[base + 1], &sBlo[base + 1], vb.y);
            split_store(&sBhi[base + 2], &sBlo[base + 2], vb.z);
            split_store(&sBhi[base + 3], &sBlo[base + 3], vb.w);
        }
        __syncthreads();

        // ---- compute: 4 K-steps of 32, 24 WMMAs each ----
        for (int kb = 0; kb < KC; kb += 32) {
            v16h ah[2], al[2], bh[4], bl[4];
            // A fragments: lane(0-15)=row, K = kb + lhi*8 + {0..7, 16..23}
            const int ka = kb + (lhi << 3);
#pragma unroll
            for (int tm = 0; tm < 2; ++tm) {
                const int rbase = (wm + tm * 16 + l16) * PITCH + ka;
                ah[tm] = ld_frag(&sAhi[rbase], &sAhi[rbase + 16]);
                al[tm] = ld_frag(&sAlo[rbase], &sAlo[rbase + 16]);
            }
            // B fragments: lane(0-15)=col, K = kb + lhi*8 + {0..7, 16..23}
            const int kbv = kb + (lhi << 3);
#pragma unroll
            for (int tn = 0; tn < 4; ++tn) {
                const int rbase = (wn + tn * 16 + l16) * PITCH + kbv;
                bh[tn] = ld_frag(&sBhi[rbase], &sBhi[rbase + 16]);
                bl[tn] = ld_frag(&sBlo[rbase], &sBlo[rbase + 16]);
            }
#pragma unroll
            for (int tm = 0; tm < 2; ++tm) {
#pragma unroll
                for (int tn = 0; tn < 4; ++tn) {
                    acc[tm][tn] = __builtin_amdgcn_wmma_f32_16x16x32_f16(
                        false, ah[tm], false, bh[tn], (short)0, acc[tm][tn], false, false);
                    acc[tm][tn] = __builtin_amdgcn_wmma_f32_16x16x32_f16(
                        false, ah[tm], false, bl[tn], (short)0, acc[tm][tn], false, false);
                    acc[tm][tn] = __builtin_amdgcn_wmma_f32_16x16x32_f16(
                        false, al[tm], false, bh[tn], (short)0, acc[tm][tn], false, false);
                }
            }
        }
    }

    // ---- fused epilogue: 0.5*exp(-(na+nb-2ab)) + 0.5*ab, NT stores ----
    // Output (268MB) streams once and must not evict a/b (reused via L2).
#pragma unroll
    for (int tm = 0; tm < 2; ++tm) {
#pragma unroll
        for (int tn = 0; tn < 4; ++tn) {
            const int gj = rowB0 + wn + tn * 16 + l16;   // col (C/D: N = lane&15)
            const float nb = sqb[gj];
#pragma unroll
            for (int r = 0; r < 8; ++r) {
                const int gi = rowA0 + wm + tm * 16 + (lhi << 3) + r;  // M = r + 8*(lane>>4)
                const float ab = acc[tm][tn][r];
                const float na = sqa[gi];
                const float sq = na + nb - 2.0f * ab;    // GAMMA = 1
                const float v  = 0.5f * __expf(-sq) + 0.5f * ab;  // ALPHA = 0.5
                __builtin_nontemporal_store(v, &out[(size_t)gi * mcols + gj]);
            }
        }
    }
}

// ---------------------------------------------------------------------------
extern "C" void kernel_launch(void* const* d_in, const int* in_sizes, int n_in,
                              void* d_out, int out_size, void* d_ws, size_t ws_size,
                              hipStream_t stream) {
    const float* a = (const float*)d_in[0];
    const float* b = (const float*)d_in[1];
    // d_in[2] (Wq) and d_in[3] (Wk) are mathematically dead: softmax over a
    // singleton axis is exactly 1, so attn_sim == a.b — never read them.
    (void)n_in; (void)out_size; (void)ws_size;

    const int n = in_sizes[0] / EMBED;   // 8192
    const int m = in_sizes[1] / EMBED;   // 8192

    float* sqa = (float*)d_ws;
    float* sqb = sqa + n;

    rownorm_kernel<<<dim3((n + 7) / 8), 256, 0, stream>>>(a, sqa, n);
    rownorm_kernel<<<dim3((m + 7) / 8), 256, 0, stream>>>(b, sqb, m);

    dim3 grid(m / BN, n / BM);
    hybrid_gram_kernel<<<grid, 256, 0, stream>>>(a, b, sqa, sqb, (float*)d_out, n, m);
}
